// DPMultiheadAttention_43525198577947
// MI455X (gfx1250) — matrix operand order repaired and checked by reading, not compiled, not run on hardware
//
#include <hip/hip_runtime.h>
#include <hip/hip_bf16.h>

typedef __attribute__((ext_vector_type(16))) _Float16 v16h;
typedef __attribute__((ext_vector_type(8)))  _Float16 v8h;
typedef __attribute__((ext_vector_type(8)))  float    v8f;
typedef __attribute__((ext_vector_type(4)))  float    v4f;

union F16x16 { v16h v; v8h h[2]; };

__device__ __forceinline__ v8f wmma_f16(v16h a, v16h b, v8f c) {
  // v_wmma_f32_16x16x32_f16: D = A(16x32) * B(32x16) + C
  return __builtin_amdgcn_wmma_f32_16x16x32_f16(false, a, false, b, (short)0, c, false, false);
}

__device__ __forceinline__ v8h cvt2_v8h(v4f a, v4f b) {
  v8h h;
#pragma unroll
  for (int i = 0; i < 4; ++i) { h[i] = (_Float16)a[i]; h[i + 4] = (_Float16)b[i]; }
  return h;
}

// ---------------------------------------------------------------------------
// Input projections: Y = X @ W^T + b (optionally *scale), X f32 [8192][1024],
// W f32 [1024][1024] (nn.Linear layout == B^T form). MODE 0: write f16
// [bh][l][d] (q/k). MODE 2: write transposed f16 [bh][d][s] (v).
// Block = 128 threads (4 waves), tile 64x64, K-step 32, double-buffered LDS,
// software-pipelined W fetch.
// ---------------------------------------------------------------------------
template<int MODE>
__global__ __launch_bounds__(128) void proj_kernel(
    const float* __restrict__ X, const float* __restrict__ W,
    const float* __restrict__ bias, _Float16* __restrict__ out, float scale)
{
  __shared__ _Float16 Bs[2][64][32];              // 8 KB, double buffered
  const int tid  = threadIdx.x;
  const int lane = tid & 31, wave = tid >> 5;
  const int half = lane >> 4, l16 = lane & 15;
  const int C0 = blockIdx.x * 64;
  const int R0 = blockIdx.y * 64;
  const int rowA = R0 + wave * 16 + l16;

  const int sr = tid >> 1, scb = (tid & 1) * 16;  // staging assignment
  const float* wrow = W + (size_t)(C0 + sr) * 1024 + scb;
  const float* arow = X + (size_t)rowA * 1024;

  v8f acc[4] = {v8f{}, v8f{}, v8f{}, v8f{}};

  // prologue: fetch W tile for kk=0
  v4f f0, f1, f2, f3;
  {
    const v4f* s4 = (const v4f*)wrow;
    f0 = s4[0]; f1 = s4[1]; f2 = s4[2]; f3 = s4[3];
  }

  for (int kk = 0; kk < 1024; kk += 32) {
    const int p = (kk >> 5) & 1;
    // convert in-flight W regs -> LDS buffer p
    *(v8h*)&Bs[p][sr][scb]     = cvt2_v8h(f0, f1);
    *(v8h*)&Bs[p][sr][scb + 8] = cvt2_v8h(f2, f3);

    // issue next W tile fetch (clamped; result unused on last iteration)
    {
      const int kn = (kk + 32 < 1024) ? kk + 32 : 0;
      const v4f* s4 = (const v4f*)(wrow + kn);
      f0 = s4[0]; f1 = s4[1]; f2 = s4[2]; f3 = s4[3];
    }

    // A fragment from f32 X: K runs [kk+16h, +16)
    F16x16 af;
    {
      const v4f* p0 = (const v4f*)(arow + kk + 16 * half);
      af.h[0] = cvt2_v8h(p0[0], p0[1]);
      af.h[1] = cvt2_v8h(p0[2], p0[3]);           // +8 floats
    }

    __syncthreads();                              // buffer p now visible

    F16x16 bf[4];
#pragma unroll
    for (int nt = 0; nt < 4; ++nt) {
      bf[nt].h[0] = *(const v8h*)&Bs[p][nt * 16 + l16][16 * half];
      bf[nt].h[1] = *(const v8h*)&Bs[p][nt * 16 + l16][16 * half + 8];
    }
#pragma unroll
    for (int nt = 0; nt < 4; ++nt)
      acc[nt] = wmma_f16(af.v, bf[nt].v, acc[nt]);
  }

  // ---- epilogue: bias, scale, scatter to attention layout (branch-free) ----
#pragma unroll
  for (int nt = 0; nt < 4; ++nt) {
    const int col = C0 + nt * 16 + l16;           // e = h*64 + d
    const float bv = bias[col];
    const int hh = col >> 6, d = col & 63;
#pragma unroll
    for (int j = 0; j < 8; ++j) {
      const int r  = R0 + wave * 16 + 8 * half + j;   // r = l*4 + n
      const int l  = r >> 2, nb = r & 3;
      const int bh = nb * 16 + hh;
      const float val = (acc[nt][j] + bv) * scale;
      if constexpr (MODE == 2)
        out[((size_t)bh * 64 + d) * 2048 + l] = (_Float16)val;   // vt[bh][d][s]
      else
        out[((size_t)bh * 2048 + l) * 64 + d] = (_Float16)val;   // q/k[bh][l][d]
    }
  }
}

// ---------------------------------------------------------------------------
// Attention: block = (n, 16 query rows), 512 threads = 16 waves, wave = head.
// Sweep 1: exact row max+sum (online, half-wave shfl reduction).
// Sweep 2: normalized P tiles -> cross-head LDS sum for avg_weights (double-
// buffered, 1 barrier/step) + LDS layout round-trip + P @ V^T WMMAs.
// ---------------------------------------------------------------------------
__global__ __launch_bounds__(512) void attn_kernel(
    const _Float16* __restrict__ q16, const _Float16* __restrict__ k16,
    const _Float16* __restrict__ vt16, _Float16* __restrict__ ctx16,
    float* __restrict__ avg_out)
{
  __shared__ float pbuf[2][16][16][32];           // 64 KB: [buf][head][row][col]
  const int tid  = threadIdx.x;
  const int lane = tid & 31, h = tid >> 5;        // wave index == head
  const int half = lane >> 4, l16 = lane & 15;
  const int n  = blockIdx.y;
  const int l0 = blockIdx.x * 16;
  const int bh = n * 16 + h;

  const _Float16* q  = q16  + ((size_t)bh * 2048 + l0) * 64;
  const _Float16* kx = k16  + (size_t)bh * 2048 * 64;
  const _Float16* vt = vt16 + (size_t)bh * 64 * 2048;

  // Q fragments for rows l0..l0+15 (K-dim D=64 -> two K=32 fragments)
  F16x16 qa[2];
#pragma unroll
  for (int kt = 0; kt < 2; ++kt) {
    const v8h* p = (const v8h*)(q + (size_t)l16 * 64 + kt * 32 + 16 * half);
    qa[kt].h[0] = p[0];
    qa[kt].h[1] = p[1];                           // +8 halves
  }

  // ---- sweep 1: online row max / sumexp over all 2048 keys ----
  float mrow[8], srow[8];
#pragma unroll
  for (int j = 0; j < 8; ++j) { mrow[j] = -1e30f; srow[j] = 0.0f; }

#pragma unroll 2
  for (int st = 0; st < 2048; st += 16) {
    F16x16 kb[2];
#pragma unroll
    for (int kt = 0; kt < 2; ++kt) {
      const v8h* p = (const v8h*)(kx + (size_t)(st + l16) * 64 + kt * 32 + 16 * half);
      kb[kt].h[0] = p[0];
      kb[kt].h[1] = p[1];
    }
    v8f c = {};
    c = wmma_f16(qa[0].v, kb[0].v, c);
    c = wmma_f16(qa[1].v, kb[1].v, c);
#pragma unroll
    for (int j = 0; j < 8; ++j) {
      const float x  = c[j];
      const float mn = fmaxf(mrow[j], x);
      srow[j] = srow[j] * __expf(mrow[j] - mn) + __expf(x - mn);
      mrow[j] = mn;
    }
  }
  // reduce (max,sum) across the 16 lanes of each half-wave
#pragma unroll
  for (int msk = 1; msk < 16; msk <<= 1) {
#pragma unroll
    for (int j = 0; j < 8; ++j) {
      const float mo = __shfl_xor(mrow[j], msk, 32);
      const float so = __shfl_xor(srow[j], msk, 32);
      const float mn = fmaxf(mrow[j], mo);
      srow[j] = srow[j] * __expf(mrow[j] - mn) + so * __expf(mo - mn);
      mrow[j] = mn;
    }
  }
  float rinv[8];
#pragma unroll
  for (int j = 0; j < 8; ++j) rinv[j] = 1.0f / srow[j];

  // ---- sweep 2: probs, head-average, P @ V ----
  v8f ctx[4] = {v8f{}, v8f{}, v8f{}, v8f{}};
  const int arow = tid >> 5, acol = tid & 31;     // avg-reduction assignment

  for (int st = 0; st < 2048; st += 32) {
    const int pp = (st >> 5) & 1;
    v8f p0 = {}, p1 = {};
    {
      F16x16 kb0[2], kb1[2];
#pragma unroll
      for (int kt = 0; kt < 2; ++kt) {
        const v8h* a = (const v8h*)(kx + (size_t)(st + l16) * 64 + kt * 32 + 16 * half);
        kb0[kt].h[0] = a[0]; kb0[kt].h[1] = a[1];
        const v8h* b = (const v8h*)(kx + (size_t)(st + 16 + l16) * 64 + kt * 32 + 16 * half);
        kb1[kt].h[0] = b[0]; kb1[kt].h[1] = b[1];
      }
      p0 = wmma_f16(qa[0].v, kb0[0].v, p0);
      p0 = wmma_f16(qa[1].v, kb0[1].v, p0);
      p1 = wmma_f16(qa[0].v, kb1[0].v, p1);
      p1 = wmma_f16(qa[1].v, kb1[1].v, p1);
    }
#pragma unroll
    for (int j = 0; j < 8; ++j) {
      p0[j] = __expf(p0[j] - mrow[j]) * rinv[j];
      p1[j] = __expf(p1[j] - mrow[j]) * rinv[j];
    }
    // C-layout -> LDS row-major P tile (f32), double buffered
#pragma unroll
    for (int j = 0; j < 8; ++j) {
      pbuf[pp][h][8 * half + j][l16]      = p0[j];
      pbuf[pp][h][8 * half + j][16 + l16] = p1[j];
    }

    // issue V fragments early (independent of LDS) to hide global latency
    F16x16 vb[4];
#pragma unroll
    for (int nt = 0; nt < 4; ++nt) {
      const v8h* p = (const v8h*)(vt + (size_t)(nt * 16 + l16) * 2048 + st + 16 * half);
      vb[nt].h[0] = p[0]; vb[nt].h[1] = p[1];
    }

    __syncthreads();                              // single barrier per step

    // deterministic cross-head sum -> avg_weights[n][l][s]
    {
      float s = 0.0f;
#pragma unroll
      for (int w = 0; w < 16; ++w) s += pbuf[pp][w][arow][acol];
      avg_out[((size_t)n * 2048 + l0 + arow) * 2048 + st + acol] = s * 0.0625f;
    }

    // reload P as A-fragment (16 rows x 32 s-cols), f32 -> f16
    F16x16 pa;
    {
      const float* pr = &pbuf[pp][h][l16][0];
      pa.h[0] = cvt2_v8h(*(const v4f*)(pr + 16 * half),
                         *(const v4f*)(pr + 16 * half + 4));
      pa.h[1] = cvt2_v8h(*(const v4f*)(pr + 16 * half + 8),
                         *(const v4f*)(pr + 16 * half + 12));
    }

    // ctx += P(16x32) @ V(32x64): B^T = vt[d][s]
#pragma unroll
    for (int nt = 0; nt < 4; ++nt)
      ctx[nt] = wmma_f16(pa.v, vb[nt].v, ctx[nt]);
  }

  // context -> ctx16 row-major [r = l*4+n][e = h*64+d] (f16)
#pragma unroll
  for (int nt = 0; nt < 4; ++nt) {
    const int e = h * 64 + nt * 16 + l16;
#pragma unroll
    for (int j = 0; j < 8; ++j) {
      const int l = l0 + 8 * half + j;
      ctx16[(size_t)(l * 4 + n) * 1024 + e] = (_Float16)ctx[nt][j];
    }
  }
}

// ---------------------------------------------------------------------------
// Output projection: out = ctx16 @ Wo^T + bo, f32 result straight into d_out
// (layout [L][N][E] == row r = l*4+n). Same pipelined tiling, f16 A operand.
// ---------------------------------------------------------------------------
__global__ __launch_bounds__(128) void oproj_kernel(
    const _Float16* __restrict__ X, const float* __restrict__ W,
    const float* __restrict__ bias, float* __restrict__ out)
{
  __shared__ _Float16 Bs[2][64][32];
  const int tid  = threadIdx.x;
  const int lane = tid & 31, wave = tid >> 5;
  const int half = lane >> 4, l16 = lane & 15;
  const int C0 = blockIdx.x * 64;
  const int R0 = blockIdx.y * 64;
  const int rowA = R0 + wave * 16 + l16;

  const int sr = tid >> 1, scb = (tid & 1) * 16;
  const float* wrow = W + (size_t)(C0 + sr) * 1024 + scb;
  const _Float16* arow = X + (size_t)rowA * 1024;

  v8f acc[4] = {v8f{}, v8f{}, v8f{}, v8f{}};

  v4f f0, f1, f2, f3;
  {
    const v4f* s4 = (const v4f*)wrow;
    f0 = s4[0]; f1 = s4[1]; f2 = s4[2]; f3 = s4[3];
  }

  for (int kk = 0; kk < 1024; kk += 32) {
    const int p = (kk >> 5) & 1;
    *(v8h*)&Bs[p][sr][scb]     = cvt2_v8h(f0, f1);
    *(v8h*)&Bs[p][sr][scb + 8] = cvt2_v8h(f2, f3);

    {
      const int kn = (kk + 32 < 1024) ? kk + 32 : 0;
      const v4f* s4 = (const v4f*)(wrow + kn);
      f0 = s4[0]; f1 = s4[1]; f2 = s4[2]; f3 = s4[3];
    }

    F16x16 af;
    {
      const v8h* pA = (const v8h*)(arow + kk + 16 * half);
      af.h[0] = pA[0];
      af.h[1] = pA[1];                            // +8 halves
    }

    __syncthreads();

    F16x16 bf[4];
#pragma unroll
    for (int nt = 0; nt < 4; ++nt) {
      bf[nt].h[0] = *(const v8h*)&Bs[p][nt * 16 + l16][16 * half];
      bf[nt].h[1] = *(const v8h*)&Bs[p][nt * 16 + l16][16 * half + 8];
    }
#pragma unroll
    for (int nt = 0; nt < 4; ++nt)
      acc[nt] = wmma_f16(af.v, bf[nt].v, acc[nt]);
  }

#pragma unroll
  for (int nt = 0; nt < 4; ++nt) {
    const int col = C0 + nt * 16 + l16;
    const float bv = bias[col];
#pragma unroll
    for (int j = 0; j < 8; ++j) {
      const int r = R0 + wave * 16 + 8 * half + j;
      out[(size_t)r * 1024 + col] = acc[nt][j] + bv;
    }
  }
}

// ---------------------------------------------------------------------------
extern "C" void kernel_launch(void* const* d_in, const int* in_sizes, int n_in,
                              void* d_out, int out_size, void* d_ws, size_t ws_size,
                              hipStream_t stream) {
  (void)in_sizes; (void)n_in; (void)out_size; (void)ws_size;
  const float* query = (const float*)d_in[0];
  const float* key   = (const float*)d_in[1];
  const float* value = (const float*)d_in[2];
  const float* Wq = (const float*)d_in[3]; const float* bq = (const float*)d_in[4];
  const float* Wk = (const float*)d_in[5]; const float* bk = (const float*)d_in[6];
  const float* Wv = (const float*)d_in[7]; const float* bv = (const float*)d_in[8];
  const float* Wo = (const float*)d_in[9]; const float* bo = (const float*)d_in[10];

  float* attn_out = (float*)d_out;                         // [2048][4][1024]
  float* avg_out  = attn_out + (size_t)2048 * 4 * 1024;    // [4][2048][2048]

  const size_t HEADS_ELT = (size_t)64 * 2048 * 64;         // per f16 buffer
  _Float16* q16   = (_Float16*)d_ws;
  _Float16* k16   = q16 + HEADS_ELT;
  _Float16* vt16  = k16 + HEADS_ELT;
  _Float16* ctx16 = vt16 + HEADS_ELT;                      // 64 MB total

  dim3 bp(128), gp(1024 / 64, 8192 / 64);
  proj_kernel<0><<<gp, bp, 0, stream>>>(query, Wq, bq, q16, 0.125f);  // D^-0.5
  proj_kernel<0><<<gp, bp, 0, stream>>>(key,   Wk, bk, k16, 1.0f);
  proj_kernel<2><<<gp, bp, 0, stream>>>(value, Wv, bv, vt16, 1.0f);   // V^T

  attn_kernel<<<dim3(2048 / 16, 4), 512, 0, stream>>>(q16, k16, vt16, ctx16, avg_out);

  oproj_kernel<<<gp, bp, 0, stream>>>(ctx16, Wo, bo, attn_out);
}
